// MSDeformAttn_SingleLevel_21414706938673
// MI455X (gfx1250) — compile-verified, matrix operand order checked by reading, not run on hardware
//
#include <hip/hip_runtime.h>

// ---------------------------------------------------------------------------
// MS-Deformable-Attention (single level) for gfx1250 (MI455X), wave32.
// GEMMs: v_wmma_f32_16x16x32_bf16 (fp32 accumulate), B tiles via TDM
// (tensor_load_to_lds) when available, vectorized manual staging otherwise.
// ---------------------------------------------------------------------------

typedef __attribute__((ext_vector_type(16))) __bf16       v16bf;
typedef __attribute__((ext_vector_type(8)))  float        v8f;
typedef __attribute__((ext_vector_type(4)))  float        f32x4;
typedef __attribute__((ext_vector_type(4)))  unsigned int u32x4;
typedef __attribute__((ext_vector_type(2)))  unsigned int u32x2;
typedef __attribute__((ext_vector_type(8)))  int          i32x8;
typedef __attribute__((ext_vector_type(4)))  int          i32x4;

#if defined(__gfx1250__) && __has_builtin(__builtin_amdgcn_tensor_load_to_lds)
#define HAVE_TDM 1
#else
#define HAVE_TDM 0
#endif

union FragBF {
    v16bf bf;
    u32x4 q[2];
};

// round-to-nearest-even fp32 -> bf16 bit pattern
__device__ __forceinline__ unsigned short f2bf(float f) {
    unsigned u = __float_as_uint(f);
    unsigned r = ((u >> 16) & 1u) + 0x7FFFu;
    return (unsigned short)((u + r) >> 16);
}
__device__ __forceinline__ unsigned pk2(float a, float b) {
    return (unsigned)f2bf(a) | ((unsigned)f2bf(b) << 16);
}

#if HAVE_TDM
// Issue one TDM 2D-tile load: Wt[0:ncols][k0:k0+32] (bf16, row stride 256)
// into LDS at lds_off with 16B pad after every 64B row -> matches LDT=40.
__device__ __forceinline__ void tdm_load_wt_tile(const unsigned short* Wt, int k0,
                                                 unsigned lds_off, int ncols)
{
    unsigned long long ga = (unsigned long long)Wt + (unsigned long long)(k0 * 2);
    u32x4 g0;
    g0[0] = 1u;                                           // count=1, user mode
    g0[1] = lds_off;                                      // lds_addr (bytes)
    g0[2] = (unsigned)(ga & 0xFFFFFFFFu);                 // global_addr[31:0]
    g0[3] = (unsigned)((ga >> 32) & 0x01FFFFFFu) | (2u << 30);  // addr[56:32] | type=2
    i32x8 g1;
    g1[0] = (int)((1u << 16)      // data_size = 1 (2 bytes)
                | (1u << 20)      // pad_enable
                | (3u << 22)      // pad_interval: 16 DWORDs (64B) per row
                | (3u << 25));    // pad_amount:   4 DWORDs (16B)
    g1[1] = (int)(256u << 16);    // tensor_dim0 = 256 (bits 79:48, low half here)
    g1[2] = (int)((unsigned)ncols << 16); // dim0 hi16=0 | tensor_dim1 lo16 = ncols
    g1[3] = (int)(32u << 16);     // tensor_dim1 hi16=0 | tile_dim0 = 32
    g1[4] = (int)(unsigned)ncols; // tile_dim1 = ncols | tile_dim2 = 0
    g1[5] = 256;                  // tensor_dim0_stride lo32 = 256 elements
    g1[6] = 0;                    // stride hi | dim1_stride lo16 (unused, 2D)
    g1[7] = 0;
    i32x4 gz = {};
#if __clang_major__ >= 23
    i32x8 gz8 = {};
    __builtin_amdgcn_tensor_load_to_lds(g0, g1, gz, gz, gz8, 0);
#else
    __builtin_amdgcn_tensor_load_to_lds(g0, g1, gz, gz, 0);
#endif
}
#endif

// ---------------------------------------------------------------------------
// One-shot: transpose + convert weights to bf16, Wt[n][k] (row stride 256).
// Single-matrix case: pass W2==W1, nc1==ncols. Fused case: [W_off | W_attn].
// grid.x == ncols, 256 threads (thread k).
// ---------------------------------------------------------------------------
__global__ __launch_bounds__(256) void wt_transpose_kernel(
    const float* __restrict__ W1, int nc1,
    const float* __restrict__ W2, int nc2,
    unsigned short* __restrict__ out)
{
    int n = blockIdx.x;
    int k = threadIdx.x;
    float v = (n < nc1) ? W1[(size_t)k * nc1 + n] : W2[(size_t)k * nc2 + (n - nc1)];
    out[(size_t)n * 256 + k] = f2bf(v);
}

// ---------------------------------------------------------------------------
// Tiled bf16 WMMA GEMM:  out = A(80000 x 256) * W(256 x NCOLS) + bias
//   MODE 0: NCOLS=256, scatter into value layout (N*heads, 10000, 32)
//   MODE 1: NCOLS=256, plain row-major (final output projection)
//   MODE 2: NCOLS=96,  bias = [b_off | b_attn], row-major out
// Block: 256 threads (8 waves), MT=64 rows. NCOLS=256: wave = (mb:2)x(nb:4),
// 32 rows x 64 cols per wave -> 8 WMMA / K-step. NCOLS=96: (mb:4)x(nb:2),
// 16 rows x 48 cols -> 3 WMMA / K-step.
// ---------------------------------------------------------------------------
template <int MODE>
__global__ __launch_bounds__(256) void gemm_bf16_wmma(
    const float* __restrict__ A,
    const unsigned short* __restrict__ Wt,   // (NCOLS, 256) bf16 pre-transposed
    const float* __restrict__ bias1,
    const float* __restrict__ bias2,         // only MODE 2
    float* __restrict__ out)
{
    constexpr int NCOLS = (MODE == 2) ? 96 : 256;
    constexpr int MT    = 64;
    constexpr int FN    = (MODE == 2) ? 3 : 4;  // 16-col frags per wave
    constexpr int AM    = (MODE == 2) ? 1 : 2;  // 16-row frags per wave
    constexpr int KSTEP = 32;
    constexpr int LDT   = 40;  // padded K-stride (ushorts): 64B data + 16B pad

    __shared__ unsigned short sB[NCOLS * LDT];
    __shared__ unsigned short sA[MT * LDT];

    const int tid  = threadIdx.x;
    const int wave = tid >> 5;
    const int lane = tid & 31;
    const int lrow = lane & 15;
    const int hl   = lane >> 4;

    int mb, nb0;
    if (NCOLS == 256) { mb = wave >> 2; nb0 = (wave & 3) * FN; }
    else              { mb = wave >> 1; nb0 = (wave & 1) * FN; }

    const int m0 = blockIdx.x * MT;

    v8f acc[AM][FN] = {};

    for (int k0 = 0; k0 < 256; k0 += KSTEP) {
        __syncthreads();  // previous iteration's LDS reads complete

#if HAVE_TDM
        if (tid == 0) {   // one TDM op per block per K-step (EXEC ignored by TDM)
            tdm_load_wt_tile(Wt, k0, (unsigned)(size_t)(void*)sB, NCOLS);
        }
#else
        // manual vectorized B staging: contiguous bf16 rows, b128 copies
        for (int e = tid; e < NCOLS * 4; e += 256) {
            int n = e >> 2, k8 = (e & 3) * 8;
            *(u32x4*)(sB + n * LDT + k8) =
                *(const u32x4*)(Wt + (size_t)n * 256 + k0 + k8);
        }
#endif
        // ---- stage A tile (MT x 32 fp32 -> packed bf16), b128 loads ----
#pragma unroll
        for (int j = 0; j < (MT * 8) / 256; ++j) {
            int cidx = tid + j * 256;
            int r = cidx >> 3, c4 = (cidx & 7) * 4;
            f32x4 v = *(const f32x4*)(A + (size_t)(m0 + r) * 256 + k0 + c4);
            u32x2 pkd;
            pkd[0] = pk2(v[0], v[1]);
            pkd[1] = pk2(v[2], v[3]);
            *(u32x2*)(sA + r * LDT + c4) = pkd;
        }
        if (k0 + KSTEP < 256) {
            __builtin_prefetch(&A[(size_t)(m0 + (tid % MT)) * 256 + k0 + KSTEP], 0, 3);
        }
#if HAVE_TDM
        __builtin_amdgcn_s_wait_tensorcnt(0);  // per-wave counter: no-op for waves 1..7
#endif
        __syncthreads();

        // ---- fragments: lane<16 -> K{0..7,16..23}; lane>=16 -> K{8..15,24..31}
        FragBF a[AM];
#pragma unroll
        for (int am = 0; am < AM; ++am) {
            int base = (mb * 16 * AM + am * 16 + lrow) * LDT + hl * 8;
            a[am].q[0] = *(const u32x4*)(sA + base);       // ds_load_b128
            a[am].q[1] = *(const u32x4*)(sA + base + 16);
        }
#pragma unroll
        for (int f = 0; f < FN; ++f) {
            FragBF b;
            int base = ((nb0 + f) * 16 + lrow) * LDT + hl * 8;
            b.q[0] = *(const u32x4*)(sB + base);
            b.q[1] = *(const u32x4*)(sB + base + 16);
#pragma unroll
            for (int am = 0; am < AM; ++am) {
                acc[am][f] = __builtin_amdgcn_wmma_f32_16x16x32_bf16(
                    false, a[am].bf, false, b.bf, (short)0, acc[am][f], false, false);
            }
        }
    }

    // ---- epilogue: C/D layout: elem i -> M = hl*8 + i, N = lrow ----
#pragma unroll
    for (int f = 0; f < FN; ++f) {
        int c = (nb0 + f) * 16 + lrow;
        float bv = (MODE == 2) ? ((c < 64) ? bias1[c] : bias2[c - 64]) : bias1[c];
#pragma unroll
        for (int am = 0; am < AM; ++am) {
#pragma unroll
            for (int i = 0; i < 8; ++i) {
                int g = m0 + mb * 16 * AM + am * 16 + hl * 8 + i;
                float v = acc[am][f][i] + bv;
                if (MODE == 0) {
                    int n = g / 10000, pix = g - n * 10000;
                    int h = c >> 5, ch = c & 31;
                    out[((size_t)((n * 8 + h) * 10000 + pix)) * 32 + ch] = v;
                } else if (MODE == 1) {
                    out[(size_t)g * 256 + c] = v;
                } else {
                    out[(size_t)g * 96 + c] = v;
                }
            }
        }
    }
}

// ---------------------------------------------------------------------------
// prep: per (n, l, h): softmax over 4 points -> attn_p (transposed layout),
// sampling coords (ix, iy) stored in flat (l*32 + h*4 + p) order so the
// reference's head-mixing reshape becomes contiguous reads downstream.
// ---------------------------------------------------------------------------
__global__ __launch_bounds__(256) void prep_kernel(
    const float* __restrict__ proj,    // (80000, 96): [offsets 64 | logits 32]
    const float* __restrict__ refpts,  // (80000, 2)
    float* __restrict__ samp,          // (8, 320000, 2)
    float* __restrict__ attn_out)      // (64, 10000, 4) == attn_p
{
    int t = blockIdx.x * 256 + threadIdx.x;
    if (t >= 8 * 10000 * 8) return;
    int n = t / 80000;
    int rem = t - n * 80000;
    int l = rem >> 3, h = rem & 7;
    size_t g = (size_t)n * 10000 + l;
    const float* row = proj + g * 96;

    float rx = refpts[g * 2 + 0], ry = refpts[g * 2 + 1];
    float bx = 2.f * (2.f * rx - 1.f) - 1.f;   // 2*((2r-1) + off/100) - 1
    float by = 2.f * (2.f * ry - 1.f) - 1.f;

    f32x4 lg = *(const f32x4*)(row + 64 + h * 4);   // logits, 16B aligned
    float mx = fmaxf(fmaxf(lg[0], lg[1]), fmaxf(lg[2], lg[3]));
    float ex[4], s = 0.f;
#pragma unroll
    for (int p = 0; p < 4; ++p) { ex[p] = __expf(lg[p] - mx); s += ex[p]; }
    float inv = 1.f / s;

#pragma unroll
    for (int p = 0; p < 4; ++p) {
        float ox = row[(h * 4 + p) * 2 + 0];
        float oy = row[(h * 4 + p) * 2 + 1];
        float gx = bx + 0.02f * ox;
        float gy = by + 0.02f * oy;
        float ix = ((gx + 1.f) * 100.f - 1.f) * 0.5f;
        float iy = ((gy + 1.f) * 100.f - 1.f) * 0.5f;
        size_t si = ((size_t)n * 320000 + (size_t)l * 32 + h * 4 + p) * 2;
        samp[si + 0] = ix;
        samp[si + 1] = iy;
        attn_out[(((size_t)(n * 8 + h)) * 10000 + l) * 4 + p] = ex[p] * inv;
    }
}

// ---------------------------------------------------------------------------
// sample: one wave per (n, head-image hh, query q); lane = channel.
// Grid coords read at flat = hh*40000 + q*4 + p (quirky reshape), each corner
// gather is a coalesced 128B load (32 lanes x 4B); value fits in 192MB L2.
// ---------------------------------------------------------------------------
__global__ __launch_bounds__(256) void sample_kernel(
    const float* __restrict__ value,  // (64, 10000, 32)
    const float* __restrict__ samp,   // (8, 320000, 2)
    const float* __restrict__ attn,   // (64, 10000, 4)
    float* __restrict__ outbuf)       // (80000, 256)
{
    int wid  = blockIdx.x * 8 + (threadIdx.x >> 5);
    int lane = threadIdx.x & 31;
    int n = wid / 80000;
    int rem = wid - n * 80000;
    int hh = rem / 10000;
    int q  = rem - hh * 10000;

    const float* vbase = value + (size_t)(n * 8 + hh) * 10000 * 32;
    size_t sbase = ((size_t)n * 320000 + (size_t)hh * 40000 + (size_t)q * 4) * 2;
    size_t abase = ((size_t)(n * 8 + hh) * 10000 + q) * 4;

    f32x4 s01 = *(const f32x4*)(samp + sbase);      // ix0 iy0 ix1 iy1
    f32x4 s23 = *(const f32x4*)(samp + sbase + 4);  // ix2 iy2 ix3 iy3
    f32x4 wts = *(const f32x4*)(attn + abase);

    float acc = 0.f;
#pragma unroll
    for (int p = 0; p < 4; ++p) {
        float ix = (p < 2) ? s01[(p & 1) * 2 + 0] : s23[(p & 1) * 2 + 0];
        float iy = (p < 2) ? s01[(p & 1) * 2 + 1] : s23[(p & 1) * 2 + 1];
        float w  = wts[p];
        float fx = floorf(ix), fy = floorf(iy);
        int   x0 = (int)fx,   y0 = (int)fy;
        float wx1 = ix - fx,  wy1 = iy - fy;
        float wx0 = 1.f - wx1, wy0 = 1.f - wy1;
        float s = 0.f;
#pragma unroll
        for (int cy = 0; cy < 2; ++cy) {
#pragma unroll
            for (int cx = 0; cx < 2; ++cx) {
                int xi = x0 + cx, yi = y0 + cy;
                float cw = (cx ? wx1 : wx0) * (cy ? wy1 : wy0);
                if (xi >= 0 && xi < 100 && yi >= 0 && yi < 100) {
                    s += cw * vbase[(size_t)(yi * 100 + xi) * 32 + lane];
                }
            }
        }
        acc += w * s;
    }
    outbuf[((size_t)n * 10000 + q) * 256 + hh * 32 + lane] = acc;
}

// ---------------------------------------------------------------------------
extern "C" void kernel_launch(void* const* d_in, const int* in_sizes, int n_in,
                              void* d_out, int out_size, void* d_ws, size_t ws_size,
                              hipStream_t stream)
{
    (void)in_sizes; (void)n_in; (void)out_size; (void)ws_size;

    const float* query  = (const float*)d_in[0];   // (8,10000,256)
    const float* refpts = (const float*)d_in[1];   // (8,10000,2)
    const float* inflat = (const float*)d_in[2];   // (8,10000,256)
    /* d_in[3] = spatial_shape (100,100) -> compile-time constant */
    const float* W_off  = (const float*)d_in[4];   // (256,64)
    const float* b_off  = (const float*)d_in[5];
    const float* W_attn = (const float*)d_in[6];   // (256,32)
    const float* b_attn = (const float*)d_in[7];
    const float* W_v    = (const float*)d_in[8];   // (256,256)
    const float* b_v    = (const float*)d_in[9];
    const float* W_o    = (const float*)d_in[10];  // (256,256)
    const float* b_o    = (const float*)d_in[11];

    float* out_main = (float*)d_out;               // (8,10000,256) = 20,480,000
    float* attn_p   = out_main + (size_t)20480000; // (8,8,1,10000,4) = 2,560,000

    float* value_ws = (float*)d_ws;                       // 20,480,000 f
    float* proj_ws  = value_ws + (size_t)20480000;        //  7,680,000 f
    float* samp_ws  = proj_ws  + (size_t)7680000;         //  5,120,000 f
    float* oattn_ws = samp_ws  + (size_t)5120000;         // 20,480,000 f
    unsigned short* wt_v = (unsigned short*)(oattn_ws + (size_t)20480000);
    unsigned short* wt_o = wt_v + 256 * 256;
    unsigned short* wt_p = wt_o + 256 * 256;              // 96 x 256

    // 0) one-shot weight transpose+convert to bf16 (Wt[n][k])
    wt_transpose_kernel<<<256, 256, 0, stream>>>(W_v, 256, W_v, 256, wt_v);
    wt_transpose_kernel<<<256, 256, 0, stream>>>(W_o, 256, W_o, 256, wt_o);
    wt_transpose_kernel<<<96,  256, 0, stream>>>(W_off, 64, W_attn, 32, wt_p);

    // 1) value projection -> scattered (N*heads, HW, 32) layout
    gemm_bf16_wmma<0><<<1250, 256, 0, stream>>>(inflat, wt_v, b_v, nullptr, value_ws);
    // 2) fused offsets+attn projection (NCOLS=96)
    gemm_bf16_wmma<2><<<1250, 256, 0, stream>>>(query, wt_p, b_off, b_attn, proj_ws);
    // 3) softmax + sampling-coordinate prep (also writes attn_p output)
    prep_kernel<<<2500, 256, 0, stream>>>(proj_ws, refpts, samp_ws, attn_p);
    // 4) bilinear sampling + point-weighted sum -> (80000,256)
    sample_kernel<<<80000, 256, 0, stream>>>(value_ws, samp_ws, attn_p, oattn_ws);
    // 5) output projection -> d_out
    gemm_bf16_wmma<1><<<1250, 256, 0, stream>>>(oattn_ws, wt_o, b_o, nullptr, out_main);
}
